// lstm_23665269800991
// MI455X (gfx1250) — matrix operand order repaired and checked by reading, not compiled, not run on hardware
//
#include <hip/hip_runtime.h>
#include <hip/hip_bf16.h>

typedef __attribute__((ext_vector_type(16))) _Float16 v16h;
typedef __attribute__((ext_vector_type(8)))  _Float16 v8h;
typedef __attribute__((ext_vector_type(8)))  float    v8f;

#define TI 2048
#define BI 512
#define II 11
#define HH 33
#define GG 132   // 4*H
#define NP 144   // padded gate dim (9 tiles of 16)
#define KP 96    // padded K dim   (3 tiles of 32)
#define NTILES 9
#define KTILES 3
#define THREADS 96   // 3 waves of 32

// Branch-free activations on native v_exp_f32 / v_rcp_f32 (no OCML slow paths).
__device__ __forceinline__ float sigmoidf_(float v) {
    return __builtin_amdgcn_rcpf(1.0f + __expf(-v));
}
__device__ __forceinline__ float tanhf_(float v) {
    // tanh(x) = 1 - 2/(1+exp(2x)); saturates correctly at +/-inf, branch-free.
    return 1.0f - 2.0f * __builtin_amdgcn_rcpf(1.0f + __expf(2.0f * v));
}

// ---------------- weight prep: swizzle weights into WMMA B-fragment order ----------------
// wfrag layout: [l][kt][nt][lane][16] halves. For lane L: n = nt*16 + (L&15),
// K-range = kt*32 + ((L>>4)*16) .. +15 (ascending K across the 16 halves).
// Wcat[k][n]: k<33 -> W_hh[n][k]; 33<=k<33+in_l -> W_ih[n][k-33]; else 0.
__global__ void prep_kernel(
    const float* __restrict__ Wih0, const float* __restrict__ Whh0,
    const float* __restrict__ bih0, const float* __restrict__ bhh0,
    const float* __restrict__ Wih1, const float* __restrict__ Whh1,
    const float* __restrict__ bih1, const float* __restrict__ bhh1,
    const float* __restrict__ Wih2, const float* __restrict__ Whh2,
    const float* __restrict__ bih2, const float* __restrict__ bhh2,
    _Float16* __restrict__ wfrag, float* __restrict__ biascat)
{
    const int NFRAGLANE = 3 * KTILES * NTILES * 32; // 2592
    int gid = blockIdx.x * blockDim.x + threadIdx.x;
    if (gid < NFRAGLANE) {
        int lane = gid & 31;
        int t = gid >> 5;
        int nt = t % NTILES; t /= NTILES;
        int kt = t % KTILES;
        int l  = t / KTILES;
        const float* Wih = (l == 0) ? Wih0 : (l == 1) ? Wih1 : Wih2;
        const float* Whh = (l == 0) ? Whh0 : (l == 1) ? Whh1 : Whh2;
        const int inl = (l == 0) ? II : HH;
        const int n = nt * 16 + (lane & 15);
        const int kbase = kt * 32 + ((lane >> 4) * 16);
        _Float16* dst = wfrag + (size_t)gid * 16;
        #pragma unroll
        for (int h = 0; h < 16; ++h) {
            int k = kbase + h;
            float v = 0.0f;
            if (n < GG) {
                if (k < HH)            v = Whh[n * HH + k];
                else if (k < HH + inl) v = Wih[n * inl + (k - HH)];
            }
            dst[h] = (_Float16)v;
        }
    } else if (gid < NFRAGLANE + 3 * NP) {
        int t = gid - NFRAGLANE;
        int l = t / NP, n = t % NP;
        const float* bih = (l == 0) ? bih0 : (l == 1) ? bih1 : bih2;
        const float* bhh = (l == 0) ? bhh0 : (l == 1) ? bhh1 : bhh2;
        biascat[l * NP + n] = (n < GG) ? (bih[n] + bhh[n]) : 0.0f;
    }
}

// ---------------- fused 3-layer LSTM over the full sequence ----------------
// One workgroup per 16-batch chunk; 3 waves, wave w owns n-tiles {3w,3w+1,3w+2}.
// All 27 B-weight fragments stay register-resident across the whole time loop
// (verified in asm). cat[l][b][k] f16: k<33 = own h_{t-1}; 33.. = layer input;
// tail zero-padded once. gbuf = f32 pre-activation gates exchanged per layer-step.
__global__ void __launch_bounds__(THREADS) lstm_kernel(
    const float* __restrict__ x,             // [B, T, I]
    const _Float16* __restrict__ wfrag,
    const float* __restrict__ biascat,
    float* __restrict__ hid)                 // [3, B, H] (into d_out)
{
    __shared__ __align__(16) _Float16 cat[3][16][KP];  // 9216 B
    __shared__ float gbuf[16][NP];                     // 9216 B

    const int tid  = threadIdx.x;
    const int wave = tid >> 5;
    const int lane = tid & 31;
    const int chunk = blockIdx.x;
    const int nl    = lane & 15;     // n-in-tile, and A-matrix row m
    const int khalf = lane >> 4;     // selects K sub-range of A/C fragments

    // zero activation buffers (covers the K/N padding forever)
    for (int i = tid; i < 3 * 16 * KP; i += THREADS)
        ((_Float16*)cat)[i] = (_Float16)0.0f;

    // per-lane bias values for this wave's 3 n-tiles, per layer
    float biasreg[3][3];
    #pragma unroll
    for (int l = 0; l < 3; ++l)
        #pragma unroll
        for (int s = 0; s < 3; ++s)
            biasreg[l][s] = biascat[l * NP + (wave * 3 + s) * 16 + nl];

    // cell state held in registers: thread handles (b,j) pairs idx = qi*96 + tid
    // qi 0..4 are always fully active (480 < 528); only qi=5 is partial (tid<48).
    float cst[3][6];
    #pragma unroll
    for (int l = 0; l < 3; ++l)
        #pragma unroll
        for (int q = 0; q < 6; ++q) cst[l][q] = 0.0f;

    // x staging: threads 0..95 each own one (b,i) = (tid/11, tid%11), b 0..8 only
    const int xb = tid / II;
    const int xi = tid - xb * II;
    const float* xptr = x + ((size_t)(chunk * 16 + (xb < 16 ? xb : 0)) * TI) * II + xi;

    __syncthreads();

    for (int t = 0; t < TI; ++t) {
        if (tid < 16 * II) {
            cat[0][xb][HH + xi] = (_Float16)xptr[(size_t)t * II];
            __builtin_prefetch(xptr + (size_t)(t + 1) * II, 0, 1);  // global_prefetch
        }
        __syncthreads();

        #pragma unroll
        for (int l = 0; l < 3; ++l) {
            // ---- A fragments from LDS (16-bit A 16x32 layout) ----
            v16h afr[KTILES];
            #pragma unroll
            for (int kt = 0; kt < KTILES; ++kt) {
                const _Float16* ap = &cat[l][nl][kt * 32 + khalf * 16];
                v8h lo = *(const v8h*)(ap);
                v8h hi = *(const v8h*)(ap + 8);
                afr[kt] = __builtin_shufflevector(lo, hi,
                    0,1,2,3,4,5,6,7,8,9,10,11,12,13,14,15);
            }
            // ---- GEMM: gates = [16,96]x[96,144] via 9 WMMAs per wave ----
            #pragma unroll
            for (int s = 0; s < 3; ++s) {
                const int nt = wave * 3 + s;
                const float bv = biasreg[l][s];
                v8f acc = { bv, bv, bv, bv, bv, bv, bv, bv };
                #pragma unroll
                for (int kt = 0; kt < KTILES; ++kt) {
                    const v16h bfr = *(const v16h*)(wfrag +
                        ((size_t)(((l * KTILES + kt) * NTILES + nt) * 32 + lane)) * 16);
                    acc = __builtin_amdgcn_wmma_f32_16x16x32_f16(
                        false, afr[kt], false, bfr, (short)0, acc, false, false);
                }
                #pragma unroll
                for (int r = 0; r < 8; ++r)
                    gbuf[khalf * 8 + r][nt * 16 + nl] = acc[r];  // C layout: m=r+8*(lane>>4)
            }
            __syncthreads();

            // ---- element-wise cell update: 528 (b,j) pairs, contiguous split ----
            #pragma unroll
            for (int qi = 0; qi < 6; ++qi) {
                const int idx = qi * THREADS + tid;
                if (qi < 5 || idx < 16 * HH) {   // guard folds away for qi<5
                    const int b = idx & 15;
                    const int j = idx >> 4;
                    const float ii = sigmoidf_(gbuf[b][j]);
                    const float ff = sigmoidf_(gbuf[b][HH + j]);
                    const float g2 = tanhf_(gbuf[b][2 * HH + j]);
                    const float oo = sigmoidf_(gbuf[b][3 * HH + j]);
                    const float c  = ff * cst[l][qi] + ii * g2;
                    cst[l][qi] = c;
                    const float h = oo * tanhf_(c);
                    const _Float16 h16 = (_Float16)h;
                    cat[l][b][j] = h16;                       // own recurrence
                    if (l < 2) cat[l + 1][b][HH + j] = h16;   // feed layer above
                    if (t == TI - 1)
                        hid[((size_t)l * BI + chunk * 16 + b) * HH + j] = h;
                }
            }
            __syncthreads();
        }
    }
}

// ---------------- MLP head: out = GELU(hidden@W1.T+b1)@W2.T+b2 ----------------
__global__ void head_kernel(const float* __restrict__ hid,
                            const float* __restrict__ W1, const float* __restrict__ b1,
                            const float* __restrict__ W2, const float* __restrict__ b2,
                            float* __restrict__ out)
{
    __shared__ float h1[72];
    const int lb = blockIdx.x;          // l*B + b  (0..1535)
    const int j  = threadIdx.x;         // 0..71
    const float* hv = hid + (size_t)lb * HH;
    float s = b1[j];
    #pragma unroll
    for (int k = 0; k < HH; ++k) s += hv[k] * W1[j * HH + k];
    h1[j] = 0.5f * s * (1.0f + erff(s * 0.70710678118654752f));  // exact GELU
    __syncthreads();
    if (j < 18) {
        float o = b2[j];
        #pragma unroll
        for (int k = 0; k < 72; ++k) o += h1[k] * W2[j * 72 + k];
        out[(size_t)lb * 18 + j] = o;
    }
}

extern "C" void kernel_launch(void* const* d_in, const int* in_sizes, int n_in,
                              void* d_out, int out_size, void* d_ws, size_t ws_size,
                              hipStream_t stream)
{
    const float* x = (const float*)d_in[0];
    const float *Wih[3], *Whh[3], *bih[3], *bhh[3];
    for (int l = 0; l < 3; ++l) {
        Wih[l] = (const float*)d_in[1 + 4 * l];
        Whh[l] = (const float*)d_in[2 + 4 * l];
        bih[l] = (const float*)d_in[3 + 4 * l];
        bhh[l] = (const float*)d_in[4 + 4 * l];
    }
    const float* W1 = (const float*)d_in[13];
    const float* b1 = (const float*)d_in[14];
    const float* W2 = (const float*)d_in[15];
    const float* b2 = (const float*)d_in[16];

    _Float16* wfrag = (_Float16*)d_ws;
    float* biascat = (float*)((char*)d_ws + (size_t)3 * KTILES * NTILES * 32 * 16 * sizeof(_Float16));

    float* out = (float*)d_out;            // [3,512,18]
    float* hid = out + 3 * BI * 18;        // [3,512,33]

    prep_kernel<<<(2592 + 3 * NP + 255) / 256, 256, 0, stream>>>(
        Wih[0], Whh[0], bih[0], bhh[0],
        Wih[1], Whh[1], bih[1], bhh[1],
        Wih[2], Whh[2], bih[2], bhh[2],
        wfrag, biascat);

    lstm_kernel<<<BI / 16, THREADS, 0, stream>>>(x, wfrag, biascat, hid);

    head_kernel<<<3 * BI, 72, 0, stream>>>(hid, W1, b1, W2, b2, out);
}
